// PtrGenOutput_77824807403876
// MI455X (gfx1250) — matrix operand order repaired and checked by reading, not compiled, not run on hardware
//
#include <hip/hip_runtime.h>
#include <hip/hip_bf16.h>
#include <math.h>

// ---------------------------------------------------------------------------
// Problem constants (from reference)
// ---------------------------------------------------------------------------
constexpr int kB    = 128;
constexpr int kH    = 1024;
constexpr int kS    = 512;
constexpr int kVOUT = 50000;
constexpr int kVIN  = 32000;

// GEMM tiling: N-tile 80 (50000 = 625*80), K-tile 64, M covered by 8 waves.
constexpr int kNT   = 80;   // vocab columns per workgroup
constexpr int kKT   = 64;   // K per LDS tile (2 WMMA K-steps)

// CDNA5 WMMA vector types (probe-verified signatures)
typedef __attribute__((ext_vector_type(16))) __bf16   v16bf;
typedef __attribute__((ext_vector_type(8)))  float    v8f;
typedef __attribute__((ext_vector_type(4)))  unsigned u32x4;
typedef __attribute__((ext_vector_type(4)))  float    f32x4;

// fp32 -> bf16 round-to-nearest-even
__device__ __forceinline__ unsigned short f2bf(float f) {
  unsigned u = __float_as_uint(f);
  unsigned r = u + 0x7FFFu + ((u >> 16) & 1u);
  return (unsigned short)(r >> 16);
}
__device__ __forceinline__ unsigned pack2(float a, float b) {
  return (unsigned)f2bf(a) | ((unsigned)f2bf(b) << 16);
}

// ---------------------------------------------------------------------------
// Kernel 0: zero a float region of workspace (inpdist + rev map)
// ---------------------------------------------------------------------------
__global__ void k_zero(float* __restrict__ p, int n) {
  int i = blockIdx.x * blockDim.x + threadIdx.x;
  if (i < n) p[i] = 0.0f;
}

// ---------------------------------------------------------------------------
// Kernel 1: convert x (B x H fp32) to bf16 (vectorized: 4 per thread)
// ---------------------------------------------------------------------------
__global__ void k_x_to_bf16(const float* __restrict__ x,
                            unsigned* __restrict__ xbf2, int n4) {
  int i = blockIdx.x * blockDim.x + threadIdx.x;   // over pairs-of-pairs
  if (i < n4) {
    f32x4 f = *reinterpret_cast<const f32x4*>(x + i * 4);
    unsigned lo = pack2(f.x, f.y);
    unsigned hi = pack2(f.z, f.w);
    xbf2[i * 2 + 0] = lo;
    xbf2[i * 2 + 1] = hi;
  }
}

// ---------------------------------------------------------------------------
// Kernel 2: reverse map rev[v] = (last j with inp_to_act[j]==v) + 1, else 0
// ---------------------------------------------------------------------------
__global__ void k_revmap(const int* __restrict__ inp_to_act,
                         unsigned* __restrict__ rev) {
  int j = blockIdx.x * blockDim.x + threadIdx.x;
  if (j < kVIN) {
    int a = inp_to_act[j];
    atomicMax(&rev[a], (unsigned)(j + 1));
  }
}

// ---------------------------------------------------------------------------
// Kernel 3: per-row small ops. One block (256 thr) per batch row.
// ---------------------------------------------------------------------------
__global__ void k_row_small(const float* __restrict__ x,
                            const int*   __restrict__ inptensor,
                            const float* __restrict__ attn_scores,
                            const float* __restrict__ out_mask,
                            const float* __restrict__ cog_W,
                            const float* __restrict__ cog_b,
                            const int*   __restrict__ inp_to_act,
                            float* __restrict__ inpdist,
                            float* __restrict__ out_attn,
                            float* __restrict__ out_pog) {
  __shared__ float red[256];
  __shared__ int   s_flag;
  const int b = blockIdx.x;
  const int t = threadIdx.x;

  // ---- attn softmax over S=512 ----
  const float* as = attn_scores + (size_t)b * kS;
  float m = -INFINITY;
  for (int s = t; s < kS; s += 256) m = fmaxf(m, as[s]);
  red[t] = m; __syncthreads();
  for (int o = 128; o > 0; o >>= 1) {
    if (t < o) red[t] = fmaxf(red[t], red[t + o]);
    __syncthreads();
  }
  const float mval = red[0];
  __syncthreads();

  float sum = 0.0f;
  for (int s = t; s < kS; s += 256) sum += __expf(as[s] - mval);
  red[t] = sum; __syncthreads();
  for (int o = 128; o > 0; o >>= 1) {
    if (t < o) red[t] += red[t + o];
    __syncthreads();
  }
  const float inv = 1.0f / red[0];
  __syncthreads();

  const int* it = inptensor + (size_t)b * kS;
  for (int s = t; s < kS; s += 256) {
    float p = __expf(as[s] - mval) * inv;
    out_attn[(size_t)b * kS + s] = p;
    atomicAdd(&inpdist[(size_t)b * kVIN + it[s]], p);
  }

  // ---- cog GEMV: two dot products over H ----
  const float* xb = x + (size_t)b * kH;
  float c0 = 0.0f, c1 = 0.0f;
  for (int h = t; h < kH; h += 256) {
    float xv = xb[h];
    c0 += xv * cog_W[h];
    c1 += xv * cog_W[kH + h];
  }
  red[t] = c0; __syncthreads();
  for (int o = 128; o > 0; o >>= 1) { if (t < o) red[t] += red[t + o]; __syncthreads(); }
  const float c0tot = red[0]; __syncthreads();
  red[t] = c1; __syncthreads();
  for (int o = 128; o > 0; o >>= 1) { if (t < o) red[t] += red[t + o]; __syncthreads(); }
  const float c1tot = red[0];

  // ---- cancopy: exists j with inp_to_act[j]!=0 and out_mask[b, act]>0 ----
  if (t == 0) s_flag = 0;
  __syncthreads();
  for (int j = t; j < kVIN; j += 256) {
    int a = inp_to_act[j];
    if (a != 0 && out_mask[(size_t)b * kVOUT + a] > 0.0f) s_flag = 1;
  }
  __syncthreads();

  if (t == 0) {
    float l0 = c0tot + cog_b[0];
    float l1 = c1tot + cog_b[1] + (s_flag ? 0.0f : -INFINITY);
    float mx = fmaxf(l0, l1);
    float e0 = __expf(l0 - mx), e1 = __expf(l1 - mx);
    float s = e0 + e1;
    out_pog[b * 2 + 0] = e0 / s;
    out_pog[b * 2 + 1] = e1 / s;
  }
}

// ---------------------------------------------------------------------------
// Kernel 4: big GEMM via bf16 WMMA, double-buffered LDS.
//   gen_logits[b][v] = sum_h x[b][h]*gen_W[v][h] + gen_b[v] + log(out_mask[b][v])
// Workgroup = 8 wave32 (full M=128) x N-tile 80. Per K-tile of 64:
// stage next fp32->bf16 W tile into LDS buffer^1 while computing 10 WMMAs
// (5 N-subtiles x 2 K-halves) from buffer. One barrier per K-tile.
// Fragment layouts per CDNA5 ISA sec 7.12.2.
// ---------------------------------------------------------------------------
__global__ __launch_bounds__(256)
void k_gemm_wmma(const unsigned short* __restrict__ xbf,  // B x H bf16
                 const float* __restrict__ gen_W,          // VOUT x H fp32
                 const float* __restrict__ gen_b,
                 const float* __restrict__ out_mask,       // B x VOUT
                 float* __restrict__ gen_logits) {         // B x VOUT
  // double-buffered bf16 W tile: [buf][n in 0..79][k in 0..63] = 2 x 10 KB
  __shared__ __align__(16) unsigned short ldsW[2][kNT][kKT];

  const int n_base = blockIdx.x * kNT;              // 625 blocks
  const int wave   = threadIdx.x >> 5;              // 0..7 -> M tile
  const int lane   = threadIdx.x & 31;
  const int group  = lane >> 4;                     // half-wave (K split)
  const int nl     = lane & 15;                     // row/col within tile
  const int m_base = wave * 16;
  const int tid    = threadIdx.x;

  // --- vectorized stage of one 80x64 fp32 tile -> bf16 LDS buffer ---
  // 80*64 = 5120 floats in chunks of 8 -> 640 chunks, 256 threads: <=3 iters.
  auto stage = [&](int buf, int k0) {
    for (int c = tid; c < (kNT * kKT / 8); c += 256) {
      int row = c >> 3;            // 0..79
      int q   = c & 7;             // 8-float chunk within the 64-wide row
      const float* src = gen_W + (size_t)(n_base + row) * kH + k0 + q * 8;
      f32x4 f0 = *reinterpret_cast<const f32x4*>(src);
      f32x4 f1 = *reinterpret_cast<const f32x4*>(src + 4);
      u32x4 p;
      p.x = pack2(f0.x, f0.y);
      p.y = pack2(f0.z, f0.w);
      p.z = pack2(f1.x, f1.y);
      p.w = pack2(f1.z, f1.w);
      *reinterpret_cast<u32x4*>(&ldsW[buf][row][q * 8]) = p;
      if (k0 + kKT < kH)
        __builtin_prefetch(src + kKT, 0, 1);   // global_prefetch_b8, next K tile
    }
  };

  v8f acc[5];
#pragma unroll
  for (int t = 0; t < 5; ++t) acc[t] = v8f{};

  stage(0, 0);
  int cur = 0;
  const unsigned short* arow = xbf + (size_t)(m_base + nl) * kH;

  for (int k0 = 0; k0 < kH; k0 += kKT) {
    __syncthreads();                      // staged tile visible; prev reads done
    if (k0 + kKT < kH) stage(cur ^ 1, k0 + kKT);

#pragma unroll
    for (int kk = 0; kk < kKT; kk += 32) {
      // A fragment (16x32 bf16), ISA layout: group 0 lanes hold
      // K in [kk,kk+8) U [kk+16,kk+24); group 1 shifted by 8.
      union { u32x4 q[2]; v16bf v; } afr;
      afr.q[0] = *reinterpret_cast<const u32x4*>(arow + k0 + kk + 8 * group);
      afr.q[1] = *reinterpret_cast<const u32x4*>(arow + k0 + kk + 16 + 8 * group);

#pragma unroll
      for (int t = 0; t < 5; ++t) {
        // B fragment (32x16): lane nl = column, group g holds K = kk+8g+0..7 U kk+16+8g+0..7
        union { u32x4 q[2]; v16bf v; } bfr;
        const unsigned short* bsrc = &ldsW[cur][t * 16 + nl][kk + 8 * group];
        bfr.q[0] = *reinterpret_cast<const u32x4*>(bsrc);
        bfr.q[1] = *reinterpret_cast<const u32x4*>(bsrc + 16);
        acc[t] = __builtin_amdgcn_wmma_f32_16x16x32_bf16(
            false, afr.v, false, bfr.v, (short)0, acc[t], false, false);
      }
    }
    cur ^= 1;
  }

  // Epilogue. C/D layout: VGPR r -> M = r + 8*group, N = nl.
#pragma unroll
  for (int t = 0; t < 5; ++t) {
    const int col = n_base + t * 16 + nl;
    const float gb = gen_b[col];
#pragma unroll
    for (int r = 0; r < 8; ++r) {
      int row = m_base + r + 8 * group;
      size_t idx = (size_t)row * kVOUT + col;
      gen_logits[idx] = acc[t][r] + gb + __logf(out_mask[idx]);
    }
  }
}

// ---------------------------------------------------------------------------
// Kernel 5: per-row max / sum(exp) over V_OUT. One block per row.
// ---------------------------------------------------------------------------
__global__ void k_row_stats(const float* __restrict__ gen_logits,
                            float* __restrict__ rmax,
                            float* __restrict__ rsum) {
  __shared__ float red[256];
  const int b = blockIdx.x, t = threadIdx.x;
  const float* gl = gen_logits + (size_t)b * kVOUT;

  float m = -INFINITY;
  for (int v = t; v < kVOUT; v += 256) m = fmaxf(m, gl[v]);
  red[t] = m; __syncthreads();
  for (int o = 128; o > 0; o >>= 1) { if (t < o) red[t] = fmaxf(red[t], red[t + o]); __syncthreads(); }
  const float mval = red[0];
  __syncthreads();

  float s = 0.0f;
  for (int v = t; v < kVOUT; v += 256) s += __expf(gl[v] - mval);
  red[t] = s; __syncthreads();
  for (int o = 128; o > 0; o >>= 1) { if (t < o) red[t] += red[t + o]; __syncthreads(); }
  if (t == 0) { rmax[b] = mval; rsum[b] = red[0]; }
}

// ---------------------------------------------------------------------------
// Kernel 6: final fuse. gen_probs, ptr gather via rev, out_probs = log(mix).
// ---------------------------------------------------------------------------
__global__ void k_final(const float* __restrict__ gen_logits,
                        const float* __restrict__ rmax,
                        const float* __restrict__ rsum,
                        const unsigned* __restrict__ rev,
                        const float* __restrict__ inpdist,
                        const float* __restrict__ out_pog,
                        float* __restrict__ out_probs,
                        float* __restrict__ out_gen) {
  int i = blockIdx.x * blockDim.x + threadIdx.x;
  if (i >= kB * kVOUT) return;
  int b = i / kVOUT;
  int v = i - b * kVOUT;
  float gp = __expf(gen_logits[i] - rmax[b]) / rsum[b];
  unsigned rv = rev[v];
  float ptr = rv ? inpdist[(size_t)b * kVIN + (rv - 1)] : 0.0f;
  float pg0 = out_pog[b * 2 + 0], pg1 = out_pog[b * 2 + 1];
  out_gen[i]   = gp;
  out_probs[i] = __logf(pg0 * gp + pg1 * ptr);
}

// ---------------------------------------------------------------------------
// Launch
// ---------------------------------------------------------------------------
extern "C" void kernel_launch(void* const* d_in, const int* in_sizes, int n_in,
                              void* d_out, int out_size, void* d_ws, size_t ws_size,
                              hipStream_t stream) {
  const float* x           = (const float*)d_in[0];
  const int*   inptensor   = (const int*)  d_in[1];
  const float* attn_scores = (const float*)d_in[2];
  const float* out_mask    = (const float*)d_in[3];
  const float* gen_W       = (const float*)d_in[4];
  const float* gen_b       = (const float*)d_in[5];
  const float* cog_W       = (const float*)d_in[6];
  const float* cog_b       = (const float*)d_in[7];
  const int*   inp_to_act  = (const int*)  d_in[8];

  // workspace layout
  char* w = (char*)d_ws;
  unsigned short* xbf      = (unsigned short*)(w);                        // 256 KB
  float*          inpdist  = (float*)(w + 262144);                        // 16.4 MB
  unsigned*       rev      = (unsigned*)(w + 262144 + (size_t)kB*kVIN*4); // 200 KB
  float*          glogits  = (float*)(w + 262144 + (size_t)kB*kVIN*4 + (size_t)kVOUT*4);
  float*          rmax     = (float*)((char*)glogits + (size_t)kB*kVOUT*4);
  float*          rsum     = rmax + kB;

  // output layout (tuple concatenated flat)
  float* out       = (float*)d_out;
  float* out_probs = out;                                  // B*VOUT
  float* out_pog   = out + (size_t)kB * kVOUT;             // B*2
  float* out_gen   = out_pog + kB * 2;                     // B*VOUT
  float* out_attn  = out_gen + (size_t)kB * kVOUT;         // B*S

  // 0. zero inpdist + rev (contiguous)
  {
    int n = kB * kVIN + kVOUT;
    k_zero<<<(n + 255) / 256, 256, 0, stream>>>(inpdist, n);
  }
  // 1. x -> bf16 (vectorized by 4)
  {
    int n4 = kB * kH / 4;
    k_x_to_bf16<<<(n4 + 255) / 256, 256, 0, stream>>>(x, (unsigned*)xbf, n4);
  }
  // 2. reverse map
  k_revmap<<<(kVIN + 255) / 256, 256, 0, stream>>>(inp_to_act, rev);
  // 3. per-row small ops
  k_row_small<<<kB, 256, 0, stream>>>(x, inptensor, attn_scores, out_mask,
                                      cog_W, cog_b, inp_to_act,
                                      inpdist, out_attn, out_pog);
  // 4. WMMA GEMM -> gen_logits (+bias +log mask)
  k_gemm_wmma<<<kVOUT / kNT, 256, 0, stream>>>(xbf, gen_W, gen_b, out_mask, glogits);
  // 5. row softmax stats
  k_row_stats<<<kB, 256, 0, stream>>>(glogits, rmax, rsum);
  // 6. final fuse
  {
    int n = kB * kVOUT;
    k_final<<<(n + 255) / 256, 256, 0, stream>>>(glogits, rmax, rsum, rev,
                                                 inpdist, out_pog,
                                                 out_probs, out_gen);
  }
}
